// MRRWeightBank_18837726561083
// MI455X (gfx1250) — compile-verified, matrix operand order checked by reading, not run on hardware
//
#include <hip/hip_runtime.h>
#include <stdint.h>

// ---------------------------------------------------------------------------
// MRR weight bank:  out[b,i,w] = sum_j in[b,j,w] * W[i,j,w]
// Batched (per-wavelength) 512x32 @ 32x32 GEMM, memory-bound (~11.5us floor
// at 23.3 TB/s).  Strategy: TDM tile load -> LDS, f16 WMMA w/ f32 accum,
// precomputed fragment-swizzled weights in d_ws (L2-resident), LDS transpose
// for coalesced 128-bit stores.
// ---------------------------------------------------------------------------

typedef _Float16 v16h __attribute__((ext_vector_type(16)));
typedef float    v8f  __attribute__((ext_vector_type(8)));
typedef unsigned int u32x4 __attribute__((ext_vector_type(4)));
typedef int      i32x4 __attribute__((ext_vector_type(4)));
typedef int      i32x8 __attribute__((ext_vector_type(8)));

#define NIN   32
#define NOUT  32
#define NWL   2048
#define NB    512
#define BT    16            // batch tile (M)
#define WT    32            // wavelength tile per workgroup
#define LIN_PITCH 33        // dwords per w-row in LDS (32 + 1 TDM pad dword)
#define LIN_DWORDS (NIN * BT * LIN_PITCH)                 // 16896
#define LOUT_B_STRIDE 1060  // chosen so all 64 banks are hit on C writeback
#define LOUT_I_STRIDE 33
#define LOUT_DWORDS (15 * LOUT_B_STRIDE + 31 * LOUT_I_STRIDE + WT)  // 16955

#if defined(__has_builtin)
#if __has_builtin(__builtin_amdgcn_tensor_load_to_lds) && \
    __has_builtin(__builtin_amdgcn_s_wait_tensorcnt)
#define HAVE_TDM 1
#endif
#endif

// ---------------------------------------------------------------------------
// Kernel 1: precompute Lorentzian weights directly in WMMA B-fragment layout.
//   B is 32(K=j) x 16(N=i).  Per ISA: lanes 0-15 hold N=0..15 / K={0..7,16..23},
//   lanes 16-31 hold N=0..15 / K={8..15,24..31}; half h -> K = kbase+(h&7)+(h>=8)*16.
//   wfrag[(w*2 + ntile)*32 + lane] = 16 halves (32B) -> one coalesced v16h load.
// ---------------------------------------------------------------------------
__global__ void mrr_wfrag_kernel(const float* __restrict__ wl,
                                 v16h* __restrict__ wfrag)
{
    const int gid  = blockIdx.x * blockDim.x + threadIdx.x;   // 0 .. 131071
    const int lane = gid & 31;
    const int nt   = (gid >> 5) & 1;
    const int w    = gid >> 6;

    const float lam   = wl[w];
    const float wlmin = 1.53e-06f;
    const float wlmax = 1.57e-06f;
    const int   kbase = (lane >> 4) << 3;           // 0 or 8
    const int   n     = nt * 16 + (lane & 15);      // output index i

    v16h f;
#pragma unroll
    for (int h = 0; h < 16; ++h) {
        const int k = kbase + (h & 7) + ((h >> 3) << 4);     // input index j
        const float idx    = (float)(n * NIN + k);
        const float center = wlmin + (wlmax - wlmin) * idx * (1.0f / 1024.0f);
        const float det    = 2.0f * 15000.0f * (lam - center) / center;
        f[h] = (_Float16)(0.5f / (1.0f + det * det));
    }
    wfrag[gid] = f;
}

// ---------------------------------------------------------------------------
// Kernel 2: main batched GEMM.
//   Block: 256 threads (8 waves).  Tile: 16 batches x 32 wavelengths.
//   Phase 1: TDM 3-D tile load  in[b0:+16, 0:32, w0:+32] -> LDS [j][b][w+pad]
//   Phase 2: per (wave, wavelength): A from LDS, B from wfrag, 1 WMMA/ntile
//   Phase 3: LDS transpose -> coalesced float4 stores along w
// ---------------------------------------------------------------------------
__global__ void mrr_main_kernel(const float* __restrict__ in,
                                const v16h* __restrict__ wfrag,
                                float* __restrict__ out)
{
    extern __shared__ float lds[];
    float* lin  = lds;               // [j=32][b=16][w pitch 33] f32
    float* lout = lds + LIN_DWORDS;  // strides LOUT_B_STRIDE / LOUT_I_STRIDE

    const int tid  = threadIdx.x;
    const int lane = tid & 31;
    const int wave = tid >> 5;
    const int w0   = blockIdx.x * WT;
    const int b0   = blockIdx.y * BT;

    // ---------------- Phase 1: stage input tile into LDS -------------------
#if HAVE_TDM
    if (wave == 0) {
        const uint64_t gaddr =
            (uint64_t)(uintptr_t)(in + ((uint64_t)b0 * NIN) * NWL + w0);
        const uint32_t lds_off = (uint32_t)(uintptr_t)lin;  // LDS aperture: low 32b = offset

        u32x4 g0;
        g0[0] = 1u;                                   // count=1, user descriptor
        g0[1] = lds_off;                              // lds_addr
        g0[2] = (uint32_t)gaddr;                      // global_addr[31:0]
        g0[3] = (uint32_t)((gaddr >> 32) & 0x1FFFFFFu) | (2u << 30);  // addr[56:32] | type=2

        // dim0 = w (contiguous), dim1 = b (stride 32*2048), dim2 = j (stride 2048)
        i32x8 g1;
        g1[0] = (2 << 16)        // data_size = 4B
              | (1 << 20)        // pad_enable
              | (4 << 22)        // pad_interval: every 32 dwords (one w-row)
              | (0 << 25);       // pad_amount: 1 dword  -> pitch 33 dwords
        g1[1] = (int)((NWL & 0xFFFF) << 16);          // tensor_dim0[15:0] = 2048
        g1[2] = (int)((NB & 0xFFFF) << 16);           // tensor_dim0 hi=0 | tensor_dim1 lo = 512
        g1[3] = (int)((WT & 0xFFFF) << 16);           // tensor_dim1 hi=0 | tile_dim0 = 32 (w)
        g1[4] = (int)(BT | (NIN << 16));              // tile_dim1 = 16 (b) | tile_dim2 = 32 (j)
        g1[5] = (int)(NIN * NWL);                     // tensor_dim0_stride = 65536 (b step)
        g1[6] = (int)((NWL & 0xFFFF) << 16);          // stride0 hi=0 | tensor_dim1_stride lo = 2048
        g1[7] = 0;                                    // tensor_dim1_stride hi

        i32x4 g2;
        g2[0] = NIN;                                  // tensor_dim2 = 32 (j extent)
        g2[1] = 0;                                    // tensor_dim3 / lds_increment
        g2[2] = 0;                                    // tensor_dim2_stride lo (unused, 3-D tile)
        g2[3] = 0;                                    // stride hi | tile_dim3 = 0
        i32x4 g3 = {0, 0, 0, 0};

#if __clang_major__ >= 23
        i32x8 g4 = {0, 0, 0, 0, 0, 0, 0, 0};
        __builtin_amdgcn_tensor_load_to_lds(g0, g1, g2, g3, g4, 0);
#else
        __builtin_amdgcn_tensor_load_to_lds(g0, g1, g2, g3, 0);
#endif
        __builtin_amdgcn_s_wait_tensorcnt(0);
    }
#else
    // Fallback: coalesced float4 loads, scalar LDS stores into padded layout.
#pragma unroll
    for (int it = 0; it < 16; ++it) {
        const int idx = it * 256 + tid;      // 4096 float4 tiles
        const int w4  = idx & 7;
        const int b   = (idx >> 3) & 15;
        const int j   = idx >> 7;
        const float4 v = *(const float4*)(in +
            (((uint64_t)(b0 + b) * NIN + j) * NWL + w0 + w4 * 4));
        float* p = lin + (j * BT + b) * LIN_PITCH + w4 * 4;
        p[0] = v.x; p[1] = v.y; p[2] = v.z; p[3] = v.w;
    }
#endif
    __syncthreads();

    // ---------------- Phase 2: WMMA per wavelength --------------------------
    // A is 16(M=b) x 32(K=j): lanes 0-15 -> M=0..15,K={0..7,16..23};
    // lanes 16-31 -> M=0..15,K={8..15,24..31}.
    const int mrow  = lane & 15;
    const int kbase = (lane >> 4) << 3;   // 0 or 8
    const int bhi   = (lane >> 4) << 3;   // C layout: upper half-wave holds M+8

#pragma unroll
    for (int ww = 0; ww < 4; ++ww) {
        const int dw    = wave * 4 + ww;   // 8 waves x 4 wavelengths = 32
        const int wglob = w0 + dw;

        v16h a;
#pragma unroll
        for (int h = 0; h < 16; ++h) {
            const int k = kbase + (h & 7) + ((h >> 3) << 4);
            a[h] = (_Float16)lin[(k * BT + mrow) * LIN_PITCH + dw];
        }

#pragma unroll
        for (int nt = 0; nt < 2; ++nt) {
            const v16h b = wfrag[(wglob * 2 + nt) * 32 + lane];
            v8f c = {};
            v8f d = __builtin_amdgcn_wmma_f32_16x16x32_f16(
                false, a, false, b, (short)0, c, false, false);

            const int iloc = nt * 16 + (lane & 15);
#pragma unroll
            for (int r = 0; r < 8; ++r) {
                const int bl = r + bhi;    // VGPR r: lanes0-15 M=r, lanes16-31 M=r+8
                lout[bl * LOUT_B_STRIDE + iloc * LOUT_I_STRIDE + dw] = d[r];
            }
        }
    }
    __syncthreads();

    // ---------------- Phase 3: coalesced stores (w fastest across lanes) ----
#pragma unroll
    for (int it = 0; it < 16; ++it) {
        const int idx = it * 256 + tid;    // 4096 float4 tiles
        const int w4  = idx & 7;
        const int i   = (idx >> 3) & 31;
        const int b   = idx >> 8;
        const int base = b * LOUT_B_STRIDE + i * LOUT_I_STRIDE + w4 * 4;
        float4 v;
        v.x = lout[base + 0];
        v.y = lout[base + 1];
        v.z = lout[base + 2];
        v.w = lout[base + 3];
        *(float4*)(out + (((uint64_t)(b0 + b) * NOUT + i) * NWL + w0 + w4 * 4)) = v;
    }
}

// ---------------------------------------------------------------------------
extern "C" void kernel_launch(void* const* d_in, const int* in_sizes, int n_in,
                              void* d_out, int out_size, void* d_ws, size_t ws_size,
                              hipStream_t stream)
{
    (void)in_sizes; (void)n_in; (void)out_size; (void)ws_size;
    const float* in = (const float*)d_in[0];    // [512, 32, 2048] f32
    const float* wl = (const float*)d_in[1];    // [2048] f32
    float* out      = (float*)d_out;            // [512, 32, 2048] f32
    v16h* wfrag     = (v16h*)d_ws;              // 2048*2*32 * 32B = 4 MB

    // 1) weights -> fragment-swizzled f16 buffer (L2-resident, 4 MB)
    mrr_wfrag_kernel<<<512, 256, 0, stream>>>(wl, wfrag);

    // 2) batched GEMM: 64 wavelength-tiles x 32 batch-tiles
    const dim3 grid(NWL / WT, NB / BT);
    const size_t shmem = (size_t)(LIN_DWORDS + LOUT_DWORDS) * sizeof(float);
    mrr_main_kernel<<<grid, dim3(256), shmem, stream>>>(in, wfrag, out);
}
